// MSMOM_23828478558392
// MI455X (gfx1250) — matrix operand order repaired and checked by reading, not compiled, not run on hardware
//
#include <hip/hip_runtime.h>
#include <hip/hip_bf16.h>

typedef __attribute__((ext_vector_type(16))) _Float16 v16h;
typedef __attribute__((ext_vector_type(8)))  _Float16 v8h;
typedef __attribute__((ext_vector_type(8)))  float    v8f;
typedef __attribute__((ext_vector_type(4)))  int      v4i;

#define BETA_F      15.0f
#define LOG2E_F     1.4426950408889634f
#define LN2_F       0.6931471805599453f
#define BIG_F       10000.0f
#define EPS_F       1e-5f

// Fixed problem shape from setup_inputs(): B=4, C=64, H=W=256, k=3, 4 rates.
#define CB 64
#define HH 256
#define WW 256
#define HW 65536
#define BB 4

// CDNA5 async global->LDS path (ASYNCcnt-tracked), feature-guarded so the
// kernel compiles on toolchains that lack the builtin.
#if defined(__gfx1250__) && __has_builtin(__builtin_amdgcn_global_load_async_to_lds_b128)
#define HAVE_ASYNC_LDS 1
#else
#define HAVE_ASYNC_LDS 0
#endif

// ---------------------------------------------------------------------------
// Kernel 1: soft erosion for one dilation rate d (dil = d+1).
//   ero = -(1/b) * logsumexp_ij b*(we[c,i,j] - x[..tap..]),  OOB tap -> +BIG
// One block per image row (W=256 threads), grid (H, C, B): fully coalesced,
// x stays L2-resident (64 MiB < 192 MB L2) across the 9 taps.
// ---------------------------------------------------------------------------
__global__ __launch_bounds__(256) void soft_erode_k(
    const float* __restrict__ x, const float* __restrict__ we,
    float* __restrict__ ero, int d)
{
    const int w = threadIdx.x;
    const int h = blockIdx.x;
    const int c = blockIdx.y;
    const int b = blockIdx.z;
    const int dil = d + 1;

    const float* wp = we + (size_t)(d * CB + c) * 9;
    const float* xp = x + (((size_t)b * CB + c) << 16);   // *HW

    float s[9];
    float m = -3.0e38f;
#pragma unroll
    for (int i = 0; i < 3; ++i) {
#pragma unroll
        for (int j = 0; j < 3; ++j) {
            const int hh = h + (i - 1) * dil;
            const int ww = w + (j - 1) * dil;
            const bool in = (hh >= 0) & (hh < HH) & (ww >= 0) & (ww < WW);
            const float xv = in ? xp[hh * WW + ww] : BIG_F;   // erosion pad = +BIG
            const float t = (wp[i * 3 + j] - xv) * (BETA_F * LOG2E_F);
            s[i * 3 + j] = t;
            m = fmaxf(m, t);
        }
    }
    float sum = 0.f;
#pragma unroll
    for (int t = 0; t < 9; ++t) sum += exp2f(s[t] - m);
    const float lse2 = m + log2f(sum);
    ero[(((size_t)b * CB + c) << 16) + h * WW + w] = -lse2 * (LN2_F / BETA_F);
}

// ---------------------------------------------------------------------------
// Kernel 2: soft dilation + BN + ReLU + f16 cast into cat channel (d*64 + c).
//   dil = (1/b) * logsumexp_ij b*(wd[c,i,j] + ero[..tap..]),  OOB tap -> -BIG
// ---------------------------------------------------------------------------
__global__ __launch_bounds__(256) void soft_dilate_bn_k(
    const float* __restrict__ ero, const float* __restrict__ wd,
    const float* __restrict__ bn_g, const float* __restrict__ bn_b,
    const float* __restrict__ bn_m, const float* __restrict__ bn_v,
    _Float16* __restrict__ cat, int d)
{
    const int w = threadIdx.x;
    const int h = blockIdx.x;
    const int c = blockIdx.y;
    const int b = blockIdx.z;
    const int dil = d + 1;

    const float* wp = wd + (size_t)(d * CB + c) * 9;
    const float* ep = ero + (((size_t)b * CB + c) << 16);

    float s[9];
    float m = -3.0e38f;
#pragma unroll
    for (int i = 0; i < 3; ++i) {
#pragma unroll
        for (int j = 0; j < 3; ++j) {
            const int hh = h + (i - 1) * dil;
            const int ww = w + (j - 1) * dil;
            const bool in = (hh >= 0) & (hh < HH) & (ww >= 0) & (ww < WW);
            const float ev = in ? ep[hh * WW + ww] : -BIG_F;  // dilation pad = -BIG
            const float t = (wp[i * 3 + j] + ev) * (BETA_F * LOG2E_F);
            s[i * 3 + j] = t;
            m = fmaxf(m, t);
        }
    }
    float sum = 0.f;
#pragma unroll
    for (int t = 0; t < 9; ++t) sum += exp2f(s[t] - m);
    const float dv = (m + log2f(sum)) * (LN2_F / BETA_F);

    const int cc = d * CB + c;                 // BN params indexed [d][c]
    const float sc = bn_g[cc] * rsqrtf(bn_v[cc] + EPS_F);
    const float y  = fmaxf((dv - bn_m[cc]) * sc + bn_b[cc], 0.f);

    // cat layout: (B, 256, H, W) f16, channel = d*64 + c
    cat[(((size_t)b * (4 * CB) + cc) << 16) + h * WW + w] = (_Float16)y;
}

// ---------------------------------------------------------------------------
// Kernel 3: convert conv_w (64 x 256 f32, [o][c]) to f16 and fold the final
// BN into per-o scale/bias: scb[0..63]=g*rsqrt(v+eps), scb[64..127]=b-m*scale.
// ---------------------------------------------------------------------------
__global__ __launch_bounds__(256) void cvt_w_k(
    const float* __restrict__ wsrc, _Float16* __restrict__ wdst,
    const float* __restrict__ g, const float* __restrict__ bnb,
    const float* __restrict__ mu, const float* __restrict__ vr,
    float* __restrict__ scb)
{
    const int i = blockIdx.x * 256 + threadIdx.x;   // 16384 elements
    wdst[i] = (_Float16)wsrc[i];
    if (i < CB) {
        const float sc = g[i] * rsqrtf(vr[i] + EPS_F);
        scb[i]      = sc;
        scb[CB + i] = bnb[i] - mu[i] * sc;
    }
}

__device__ __forceinline__ unsigned pack2h(_Float16 lo, _Float16 hi)
{
    union { _Float16 h[2]; unsigned u; } p;
    p.h[0] = lo; p.h[1] = hi;
    return p.u;
}

// ---------------------------------------------------------------------------
// Kernel 4: WMMA GEMM  out[b,o,p] = relu(scale[o]*(sum_c W[o,c]*cat[b,c,p])+bias[o])
//   M = 64 (out ch), K = 256 (cat ch), N = 262144 pixels.
//
// Block: 256 threads = 8 wave32s, 4 (M-tiles) x 2 (N-tiles) -> 64 x 32 tile,
// K-loop of 8 x v_wmma_f32_16x16x32_f16 per wave.
//
// A (64x256 f16, 32 KB) is DMA'd into LDS via GLOBAL_LOAD_ASYNC_TO_LDS_B128
// (no VGPR round-trip; ASYNCcnt + s_wait_asynccnt), rows padded to 264 halfs.
// B tile (256 ch x 32 px, 16 KB) is staged once per block, transposed to
// [pixel][channel] via register pack + ds_store_b32, so every wave's B
// fragment per K-chunk is 16 contiguous halfs -> two ds_load_b128 per lane.
// Global traffic for cat is exactly one pass (8192 x 16 KB = 128 MiB).
//
// Fragment packing per CDNA5 ISA 7.12.2 (wave32):
//   A (16x32 f16): lane m = lane&15; K-halves split on lane bit4.
//   B (32x16 f16): lane n = lane&15; VGPR v covers K = lh*16 + 2v (+1): contiguous k order used here, not the ISA table's.
//   C/D (16x16 f32): N = lane&15, M = v + 8*(lane>>4).
// ---------------------------------------------------------------------------
#define BROW 264   // padded LDS row stride in halfs (528 B, 16B aligned)

__global__ __launch_bounds__(256) void gemm_bn_relu_k(
    const _Float16* __restrict__ cat, const _Float16* __restrict__ Ah,
    const float* __restrict__ scb, float* __restrict__ out)
{
    __shared__ _Float16 shA[CB * BROW];   // ~33.8 KB, padded rows of conv_w f16
    __shared__ _Float16 shB[32 * BROW];   // ~16.5 KB, [pixel][channel]

    const int tid   = threadIdx.x;
    const int wave  = tid >> 5;
    const int lane  = tid & 31;
    const int mtile = wave >> 1;          // 0..3  -> out channels [16*mtile, +16)
    const int ntile = wave & 1;           // 0..1  -> pixel sub-tile
    const int lm    = lane & 15;
    const int lh    = lane >> 4;

    const long q0 = (long)blockIdx.x * 32;        // 32 pixels per block, never
    const int  b  = (int)(q0 >> 16);              // crosses a batch boundary
    const int  p0 = (int)(q0 & (HW - 1));

    // ---------- A: async DMA global -> LDS (2048 x 16B chunks / block) ------
#pragma unroll
    for (int r = 0; r < 8; ++r) {
        const int ci  = tid + r * 256;            // chunk id
        const int row = ci >> 5;                  // 32 x 16B chunks per row
        const int c16 = ci & 31;
        const _Float16* gsrc = Ah + (size_t)row * (4 * CB) + c16 * 8;
        _Float16*       ldst = &shA[row * BROW + c16 * 8];
#if HAVE_ASYNC_LDS
        __builtin_amdgcn_global_load_async_to_lds_b128(
            (__attribute__((address_space(1))) v4i*)gsrc,
            (__attribute__((address_space(3))) v4i*)ldst,
            /*offset=*/0, /*cpol=*/0);
#else
        *(v8h*)ldst = *(const v8h*)gsrc;
#endif
    }

    // ---------- B: cooperative transposed staging of the cat tile ----------
    // thread t: channel pair c0 = 2*(t&127), pixel half ph = t>>7 (16 px).
    {
        const int cp = tid & 127;
        const int ph = tid >> 7;
        const int c0 = cp * 2;
        const _Float16* r0 =
            cat + ((size_t)b * (4 * CB) + c0) * HW + p0 + ph * 16;
        const _Float16* r1 = r0 + HW;
        const v8h x00 = *(const v8h*)(r0);
        const v8h x01 = *(const v8h*)(r0 + 8);
        const v8h x10 = *(const v8h*)(r1);
        const v8h x11 = *(const v8h*)(r1 + 8);
#pragma unroll
        for (int i = 0; i < 8; ++i) {
            const int pxa = ph * 16 + i;
            const int pxb = ph * 16 + 8 + i;
            *(unsigned*)&shB[pxa * BROW + c0] = pack2h(x00[i], x10[i]);
            *(unsigned*)&shB[pxb * BROW + c0] = pack2h(x01[i], x11[i]);
        }
    }

#if HAVE_ASYNC_LDS
#if __has_builtin(__builtin_amdgcn_s_wait_asynccnt)
    __builtin_amdgcn_s_wait_asynccnt(0);
#else
    asm volatile("s_wait_asynccnt 0x0" ::: "memory");
#endif
#endif
    __syncthreads();

    const int m  = mtile * 16 + lm;               // this lane's A row
    const int nl = ntile * 16 + lm;               // this lane's pixel (local)

    v8f acc = {};
#pragma unroll
    for (int kk = 0; kk < 4 * CB; kk += 32) {
        // ---- A fragment: 2 x ds_load_b128 from padded LDS rows ----
        const _Float16* arow = &shA[m * BROW + kk + lh * 16];
        const v8h a0 = *(const v8h*)(arow);
        const v8h a1 = *(const v8h*)(arow + 8);
        v16h af;
#pragma unroll
        for (int i = 0; i < 8; ++i) { af[i] = a0[i]; af[i + 8] = a1[i]; }

        // ---- B fragment: 16 contiguous halfs from the transposed tile ----
        const _Float16* brow = &shB[nl * BROW + kk + lh * 16];
        const v8h b0 = *(const v8h*)(brow);
        const v8h b1 = *(const v8h*)(brow + 8);
        v16h bf;
#pragma unroll
        for (int i = 0; i < 8; ++i) { bf[i] = b0[i]; bf[i + 8] = b1[i]; }

        acc = __builtin_amdgcn_wmma_f32_16x16x32_f16(
            /*neg_a=*/false, af, /*neg_b=*/false, bf,
            /*c_mod=*/(short)0, acc, /*reuse_a=*/false, /*reuse_b=*/false);
    }

    // ---- fused (pre-folded) BN + ReLU epilogue, f32 store ----
    const size_t obase = ((size_t)b * CB) * HW + p0 + ntile * 16 + lm;
#pragma unroll
    for (int v = 0; v < 8; ++v) {
        const int o = mtile * 16 + lh * 8 + v;    // C/D layout: M = v + 8*lh
        out[obase + (size_t)o * HW] = fmaxf(acc[v] * scb[o] + scb[CB + o], 0.f);
    }
}

// ---------------------------------------------------------------------------
extern "C" void kernel_launch(void* const* d_in, const int* in_sizes, int n_in,
                              void* d_out, int out_size, void* d_ws, size_t ws_size,
                              hipStream_t stream)
{
    const float* x      = (const float*)d_in[0];
    const float* we     = (const float*)d_in[1];
    const float* wd     = (const float*)d_in[2];
    const float* bn_g   = (const float*)d_in[3];
    const float* bn_b   = (const float*)d_in[4];
    const float* bn_m   = (const float*)d_in[5];
    const float* bn_v   = (const float*)d_in[6];
    const float* conv_w = (const float*)d_in[7];
    const float* bnf_g  = (const float*)d_in[8];
    const float* bnf_b  = (const float*)d_in[9];
    const float* bnf_m  = (const float*)d_in[10];
    const float* bnf_v  = (const float*)d_in[11];
    float* out = (float*)d_out;

    char* ws = (char*)d_ws;
    float*     ero = (float*)ws;                               // 64 MiB (reused per rate)
    _Float16*  cat = (_Float16*)(ws + ((size_t)64 << 20));     // 128 MiB
    _Float16*  Ah  = (_Float16*)(ws + ((size_t)192 << 20));    // 32 KiB
    float*     scb = (float*)(ws + ((size_t)192 << 20) + (64 << 10)); // 512 B

    cvt_w_k<<<64, 256, 0, stream>>>(conv_w, Ah, bnf_g, bnf_b, bnf_m, bnf_v, scb);

    const dim3 mg(HH, CB, BB);
    for (int d = 0; d < 4; ++d) {
        soft_erode_k<<<mg, 256, 0, stream>>>(x, we, ero, d);
        soft_dilate_bn_k<<<mg, 256, 0, stream>>>(ero, wd, bn_g, bn_b, bn_m, bn_v,
                                                 cat, d);
    }

    // 262144 pixels / 32 per block
    gemm_bn_relu_k<<<(BB * HW) / 32, 256, 0, stream>>>(cat, Ah, scb, out);
}
